// GGNNRelReason_5669356831749
// MI455X (gfx1250) — matrix operand order repaired and checked by reading, not compiled, not run on hardware
//
#include <hip/hip_runtime.h>
#include <hip/hip_bf16.h>

typedef __attribute__((ext_vector_type(16))) _Float16 v16h;
typedef __attribute__((ext_vector_type(8)))  _Float16 v8h;
typedef __attribute__((ext_vector_type(8)))  float    v8f;
typedef __attribute__((ext_vector_type(4)))  int      v4i;

#define HH      512
#define NRELK   51
#define NNODE   53
#define RREL    1024
#define NOBJN   2048
#define NCLSN   151
#define BIGROWS (RREL * NNODE)          /* 54272 */
#define BIGELEM ((size_t)BIGROWS * HH)  /* 27,787,264 */

// ---- CDNA5 async global->LDS path (guarded; falls back to pipelined sync) ----
#if defined(__has_builtin)
#if __has_builtin(__builtin_amdgcn_global_load_async_to_lds_b128)
#define HAVE_ASYNC_LDS 1
#endif
#endif
#ifndef HAVE_ASYNC_LDS
#define HAVE_ASYNC_LDS 0
#endif

#if HAVE_ASYNC_LDS
typedef __attribute__((address_space(1))) v4i glb_v4i;   // prints as "__device__" in HIP
typedef __attribute__((address_space(3))) v4i lds_v4i;   // prints as "__shared__"

__device__ __forceinline__ void async_cp16(const _Float16* g, _Float16* l) {
  // 16B per lane, memory -> LDS, tracked by ASYNCcnt
  __builtin_amdgcn_global_load_async_to_lds_b128(
      (glb_v4i*)(g), (lds_v4i*)(l), 0, 0);
}
__device__ __forceinline__ void wait_async_le4() {
#if __has_builtin(__builtin_amdgcn_s_wait_asynccnt)
  __builtin_amdgcn_s_wait_asynccnt(4);
#else
  asm volatile("s_wait_asynccnt 0x4" ::: "memory");
#endif
}
__device__ __forceinline__ void wait_async_le0() {
#if __has_builtin(__builtin_amdgcn_s_wait_asynccnt)
  __builtin_amdgcn_s_wait_asynccnt(0);
#else
  asm volatile("s_wait_asynccnt 0x0" ::: "memory");
#endif
}
#endif

// ======================= WMMA GEMM =======================
// C[M,N] = epi( A[M,K](f16) @ W[N,K]^T(f16) + bias )
// A split at ksplit between A1|A2 (both leading dim lda).
// EPI: 0 = f16 out, 1 = relu+f16 out, 2 = f32 out.

__device__ __forceinline__ void wmma_tile_compute(
    const _Float16* __restrict__ tA, const _Float16* __restrict__ tB,
    int wm, int wn, int lm, int lh, v8f acc[2][4])
{
  constexpr int LDT = 40;
  v16h afr[2], bfr[4];
  for (int i = 0; i < 2; ++i) {
    // A 16x32 f16: lanes0-15 e0..7->K0..7, e8..15->K16..23; lanes16-31: +8
    const _Float16* p = &tA[(wm + i * 16 + lm) * LDT];
    union { v16h v; v8h h[2]; } u;
    u.h[0] = *(const v8h*)(p + lh * 8);
    u.h[1] = *(const v8h*)(p + 16 + lh * 8);
    afr[i] = u.v;
  }
  for (int j = 0; j < 4; ++j) {
    // B 32x16 f16: column = lane&15, lanes0-15 e0..7->K0..7, e8..15->K16..23; lanes16-31: +8
    const _Float16* p = &tB[(wn + j * 16 + lm) * LDT];
    union { v16h v; v8h h[2]; } u;
    u.h[0] = *(const v8h*)(p + lh * 8);
    u.h[1] = *(const v8h*)(p + 16 + lh * 8);
    bfr[j] = u.v;
  }
  for (int i = 0; i < 2; ++i)
    for (int j = 0; j < 4; ++j)
      acc[i][j] = __builtin_amdgcn_wmma_f32_16x16x32_f16(
          false, afr[i], false, bfr[j], (short)0, acc[i][j], false, false);
}

template<int EPI>
__global__ __launch_bounds__(256)
void gemm_wmma(const _Float16* __restrict__ A1, const _Float16* __restrict__ A2,
               int ksplit, const _Float16* __restrict__ W,
               const float* __restrict__ bias, void* __restrict__ Cout,
               int M, int N, int K, int lda, int ldw, int ldc)
{
  constexpr int LDT = 40;  // half-stride padding (32 -> 40) vs LDS bank conflicts

  const int t    = threadIdx.x;
  const int lane = t & 31;
  const int wave = t >> 5;
  const int wm   = (wave >> 1) * 32;   // 4 wave rows  -> 128 M
  const int wn   = (wave &  1) * 64;   // 2 wave cols  -> 128 N
  const int lm   = lane & 15;
  const int lh   = (lane >> 4) & 1;
  const int m0   = blockIdx.y * 128;
  const int n0   = blockIdx.x * 128;

  v8f acc[2][4];
  for (int i = 0; i < 2; ++i)
    for (int j = 0; j < 4; ++j)
      for (int e = 0; e < 8; ++e) acc[i][j][e] = 0.0f;

  const int nk = K >> 5;

#if HAVE_ASYNC_LDS
  // ---------- double-buffered async DMA pipeline ----------
  __shared__ __align__(16) _Float16 sA[2][128 * LDT];
  __shared__ __align__(16) _Float16 sB[2][128 * LDT];

  // each wave issues exactly 4 async b128 ops per tile (in-order retirement)
  auto issue_tile = [&](int k0, int buf) {
    const _Float16* Asrc; int kloc;
    if (k0 < ksplit) { Asrc = A1; kloc = k0; } else { Asrc = A2; kloc = k0 - ksplit; }
    for (int c = t; c < 512; c += 256) {
      int row = c >> 2, col8 = (c & 3) << 3;
      int gr = m0 + row; if (gr >= M) gr = M - 1;   // clamp: garbage only hits unstored rows
      async_cp16(Asrc + (size_t)gr * lda + kloc + col8, &sA[buf][row * LDT + col8]);
      int gn = n0 + row; if (gn >= N) gn = N - 1;   // clamp: garbage only hits unstored cols
      async_cp16(W + (size_t)gn * ldw + k0 + col8, &sB[buf][row * LDT + col8]);
    }
  };

  issue_tile(0, 0);
  for (int i = 0; i < nk; ++i) {
    if (i + 1 < nk) {
      issue_tile((i + 1) << 5, (i + 1) & 1);  // prefetch next tile while computing
      wait_async_le4();                       // tile i landed (next tile still in flight)
    } else {
      wait_async_le0();
    }
    __syncthreads();
    wmma_tile_compute(sA[i & 1], sB[i & 1], wm, wn, lm, lh, acc);
    __syncthreads();                          // buffer reuse guard for tile i+2
  }
#else
  // ---------- register-staged pipelined sync path ----------
  __shared__ __align__(16) _Float16 sA1[128 * LDT];
  __shared__ __align__(16) _Float16 sB1[128 * LDT];

  v8h ra[2], rb[2];
  auto gload = [&](int k0) {
    const _Float16* Asrc; int kloc;
    if (k0 < ksplit) { Asrc = A1; kloc = k0; } else { Asrc = A2; kloc = k0 - ksplit; }
    for (int u = 0; u < 2; ++u) {
      int c = t + (u << 8);
      int row = c >> 2, col8 = (c & 3) << 3;
      int gr = m0 + row;
      if (gr < M) {
        ra[u] = *(const v8h*)(Asrc + (size_t)gr * lda + kloc + col8);
        if (k0 + 32 < K)
          __builtin_prefetch(Asrc + (size_t)gr * lda + kloc + col8 + 32, 0, 0);
      } else {
        for (int e = 0; e < 8; ++e) ra[u][e] = (_Float16)0.0f;
      }
      int gn = n0 + row;
      if (gn < N) {
        rb[u] = *(const v8h*)(W + (size_t)gn * ldw + k0 + col8);
      } else {
        for (int e = 0; e < 8; ++e) rb[u][e] = (_Float16)0.0f;
      }
    }
  };

  gload(0);
  for (int i = 0; i < nk; ++i) {
    for (int u = 0; u < 2; ++u) {
      int c = t + (u << 8);
      int row = c >> 2, col8 = (c & 3) << 3;
      *(v8h*)(&sA1[row * LDT + col8]) = ra[u];
      *(v8h*)(&sB1[row * LDT + col8]) = rb[u];
    }
    __syncthreads();
    if (i + 1 < nk) gload((i + 1) << 5);   // global latency overlaps the WMMAs below
    wmma_tile_compute(sA1, sB1, wm, wn, lm, lh, acc);
    __syncthreads();
  }
#endif

  // epilogue: C/D layout -> VGPR e: lanes0-15 M=e, lanes16-31 M=8+e; N=lane&15
  for (int i = 0; i < 2; ++i) {
    for (int j = 0; j < 4; ++j) {
      int gn = n0 + wn + j * 16 + lm;
      if (gn >= N) continue;
      float b = bias ? bias[gn] : 0.0f;
      for (int e = 0; e < 8; ++e) {
        int gm = m0 + wm + i * 16 + lh * 8 + e;
        if (gm >= M) continue;
        float x = acc[i][j][e] + b;
        if (EPI == 1) x = fmaxf(x, 0.0f);
        if (EPI == 2) ((float*)Cout)[(size_t)gm * ldc + gn] = x;
        else          ((_Float16*)Cout)[(size_t)gm * ldc + gn] = (_Float16)x;
      }
    }
  }
}

// ======================= small kernels =======================

__global__ void k_cvt_f16(const float* __restrict__ src, _Float16* __restrict__ dst, size_t n) {
  size_t i = (size_t)blockIdx.x * blockDim.x + threadIdx.x;
  if (i < n) dst[i] = (_Float16)src[i];
}

// fold [512,1024] -> [512,512]: Wf[n,k] = w[n,k] + w[n,k+512]  (av is tile(x,2))
__global__ void k_fold(const float* __restrict__ w, _Float16* __restrict__ wf) {
  int idx = blockIdx.x * blockDim.x + threadIdx.x;
  if (idx >= HH * HH) return;
  int n = idx >> 9, k = idx & 511;
  wf[idx] = (_Float16)(w[n * 1024 + k] + w[n * 1024 + 512 + k]);
}

__global__ void k_argmax(const float* __restrict__ logits, int* __restrict__ preds) {
  int i = blockIdx.x * blockDim.x + threadIdx.x;
  if (i >= NOBJN) return;
  const float* row = logits + (size_t)i * NCLSN;
  float best = row[1]; int bi = 1;
  for (int c = 2; c < NCLSN; ++c) { float v = row[c]; if (v > best) { best = v; bi = c; } }
  preds[i] = bi;   // argmax over [1:] then +1 == index in full row starting at 1
}

__global__ void k_gatherA(const int* __restrict__ rel_inds, const int* __restrict__ preds,
                          const float* __restrict__ prior, float* __restrict__ Arel) {
  int r = blockIdx.x, k = threadIdx.x;
  int sub = rel_inds[r * 3 + 1], obj = rel_inds[r * 3 + 2];
  int sp = preds[sub], op = preds[obj];
  if (k < NRELK) Arel[r * 64 + k] = prior[((size_t)sp * NCLSN + op) * NRELK + k];
}

__global__ void k_build_input(const _Float16* __restrict__ of, const _Float16* __restrict__ v,
                              const int* __restrict__ rel_inds,
                              _Float16* __restrict__ inflat, _Float16* __restrict__ flat) {
  int r = blockIdx.x;
  int sub = rel_inds[r * 3 + 1], obj = rel_inds[r * 3 + 2];
  const _Float16* o0 = of + (size_t)sub * HH;
  const _Float16* o1 = of + (size_t)obj * HH;
  const _Float16* vv = v + (size_t)r * HH;
  size_t base = (size_t)r * NNODE * HH;
  for (int h = threadIdx.x; h < HH; h += blockDim.x) {
    _Float16 a = o0[h], b = o1[h], c = vv[h];
    inflat[base + h] = a;           flat[base + h] = a;
    inflat[base + HH + h] = b;      flat[base + HH + h] = b;
    for (int k = 0; k < NRELK; ++k) {
      inflat[base + (size_t)(2 + k) * HH + h] = c;
      flat  [base + (size_t)(2 + k) * HH + h] = c;
    }
  }
}

// folded "av" operand: rows 0,1 = m_top = A . hidden[2:], rows 2..52 = A[k]*(h0+h1)
__global__ void k_xav(const _Float16* __restrict__ flat, const float* __restrict__ Arel,
                      _Float16* __restrict__ xav) {
  int r = blockIdx.x;
  __shared__ float As[64];
  if (threadIdx.x < NRELK) As[threadIdx.x] = Arel[r * 64 + threadIdx.x];
  __syncthreads();
  size_t base = (size_t)r * NNODE * HH;
  for (int h = threadIdx.x; h < HH; h += blockDim.x) {
    float s01 = (float)flat[base + h] + (float)flat[base + HH + h];
    float m = 0.0f;
    for (int k = 0; k < NRELK; ++k)
      m += As[k] * (float)flat[base + (size_t)(2 + k) * HH + h];
    xav[base + h]      = (_Float16)m;
    xav[base + HH + h] = (_Float16)m;
    for (int k = 0; k < NRELK; ++k)
      xav[base + (size_t)(2 + k) * HH + h] = (_Float16)(As[k] * s01);
  }
}

// z = sigmoid(zpre + gu); r = sigmoid(rpre + gu); rf = r * flat
__global__ void k_zr(const _Float16* __restrict__ zpre, const _Float16* __restrict__ rpre,
                     const _Float16* __restrict__ gu, const _Float16* __restrict__ flat,
                     _Float16* __restrict__ z_out, _Float16* __restrict__ rf_out, size_t n) {
  size_t i = (size_t)blockIdx.x * blockDim.x + threadIdx.x;
  if (i >= n) return;
  float g  = (float)gu[i];
  float z  = 1.0f / (1.0f + expf(-((float)zpre[i] + g)));
  float rr = 1.0f / (1.0f + expf(-((float)rpre[i] + g)));
  z_out[i]  = (_Float16)z;
  rf_out[i] = (_Float16)(rr * (float)flat[i]);
}

// flat = (1-z)*flat + z*tanh(hw + hu)
__global__ void k_update(const _Float16* __restrict__ z, const _Float16* __restrict__ hw,
                         const _Float16* __restrict__ hu, _Float16* __restrict__ flat, size_t n) {
  size_t i = (size_t)blockIdx.x * blockDim.x + threadIdx.x;
  if (i >= n) return;
  float zz = (float)z[i];
  float hv = tanhf((float)hw[i] + (float)hu[i]);
  flat[i] = (_Float16)((1.0f - zz) * (float)flat[i] + zz * hv);
}

__global__ void k_preds_out(const int* __restrict__ preds, float* __restrict__ out) {
  int i = blockIdx.x * blockDim.x + threadIdx.x;
  if (i < NOBJN) out[i] = (float)preds[i];
}

// ======================= host =======================
extern "C" void kernel_launch(void* const* d_in, const int* in_sizes, int n_in,
                              void* d_out, int out_size, void* d_ws, size_t ws_size,
                              hipStream_t stream)
{
  const float* obj_fmaps  = (const float*)d_in[0];
  const float* obj_logits = (const float*)d_in[1];
  const int*   rel_inds   = (const int*)  d_in[2];
  const float* vr         = (const float*)d_in[3];
  const float* prior      = (const float*)d_in[4];
  const float* obj_proj_w = (const float*)d_in[5];
  const float* obj_proj_b = (const float*)d_in[6];
  const float* rel_proj_w = (const float*)d_in[7];
  const float* rel_proj_b = (const float*)d_in[8];
  const float* eq3w_w = (const float*)d_in[9];
  const float* eq3w_b = (const float*)d_in[10];
  const float* eq3u_w = (const float*)d_in[11];
  const float* eq3u_b = (const float*)d_in[12];
  const float* eq4w_w = (const float*)d_in[13];
  const float* eq4w_b = (const float*)d_in[14];
  const float* eq5w_w = (const float*)d_in[15];
  const float* eq5w_b = (const float*)d_in[16];
  const float* eq5u_w = (const float*)d_in[17];
  const float* eq5u_b = (const float*)d_in[18];
  const float* out_w  = (const float*)d_in[19];
  const float* out_b  = (const float*)d_in[20];
  const float* cls_w  = (const float*)d_in[21];
  const float* cls_b  = (const float*)d_in[22];

  char* ws = (char*)d_ws;
  size_t off = 0;
  auto alloc = [&](size_t bytes) -> void* {
    void* p = ws + off;
    off += (bytes + 255) & ~(size_t)255;
    return p;
  };

  int*      preds   = (int*)alloc((size_t)NOBJN * 4);
  float*    Arel    = (float*)alloc((size_t)RREL * 64 * 4);
  _Float16* ofm16   = (_Float16*)alloc((size_t)NOBJN * 4096 * 2);
  _Float16* vr16    = (_Float16*)alloc((size_t)RREL * 4096 * 2);
  _Float16* opw16   = (_Float16*)alloc((size_t)HH * 4096 * 2);
  _Float16* rpw16   = (_Float16*)alloc((size_t)HH * 4096 * 2);
  _Float16* eq3u16  = (_Float16*)alloc((size_t)HH * HH * 2);
  _Float16* eq5u16  = (_Float16*)alloc((size_t)HH * HH * 2);
  _Float16* wzf     = (_Float16*)alloc((size_t)HH * HH * 2);
  _Float16* wrf     = (_Float16*)alloc((size_t)HH * HH * 2);
  _Float16* whf     = (_Float16*)alloc((size_t)HH * HH * 2);
  _Float16* outw16  = (_Float16*)alloc((size_t)HH * 1024 * 2);
  _Float16* clsw16  = (_Float16*)alloc((size_t)NRELK * (NNODE * HH) * 2);
  _Float16* of16    = (_Float16*)alloc((size_t)NOBJN * HH * 2);
  _Float16* v16     = (_Float16*)alloc((size_t)RREL * HH * 2);
  _Float16* inflat  = (_Float16*)alloc(BIGELEM * 2);
  _Float16* flat    = (_Float16*)alloc(BIGELEM * 2);
  _Float16* xav     = (_Float16*)alloc(BIGELEM * 2);   // reused as rf
  _Float16* gu      = (_Float16*)alloc(BIGELEM * 2);   // reused as hu
  _Float16* zpre    = (_Float16*)alloc(BIGELEM * 2);   // reused as z, then "out"
  _Float16* rpre    = (_Float16*)alloc(BIGELEM * 2);
  _Float16* hw      = (_Float16*)alloc(BIGELEM * 2);

  auto cvt = [&](const float* s, _Float16* d, size_t n) {
    k_cvt_f16<<<dim3((unsigned)((n + 255) / 256)), dim3(256), 0, stream>>>(s, d, n);
  };

  // ---- weight / input conversions & folds ----
  cvt(obj_fmaps,  ofm16,  (size_t)NOBJN * 4096);
  cvt(vr,         vr16,   (size_t)RREL  * 4096);
  cvt(obj_proj_w, opw16,  (size_t)HH * 4096);
  cvt(rel_proj_w, rpw16,  (size_t)HH * 4096);
  cvt(eq3u_w,     eq3u16, (size_t)HH * HH);
  cvt(eq5u_w,     eq5u16, (size_t)HH * HH);
  cvt(out_w,      outw16, (size_t)HH * 1024);
  cvt(cls_w,      clsw16, (size_t)NRELK * NNODE * HH);
  k_fold<<<dim3((HH * HH + 255) / 256), dim3(256), 0, stream>>>(eq3w_w, wzf);
  k_fold<<<dim3((HH * HH + 255) / 256), dim3(256), 0, stream>>>(eq4w_w, wrf);
  k_fold<<<dim3((HH * HH + 255) / 256), dim3(256), 0, stream>>>(eq5w_w, whf);

  // ---- preds, prior gather ----
  k_argmax<<<dim3((NOBJN + 255) / 256), dim3(256), 0, stream>>>(obj_logits, preds);
  k_gatherA<<<dim3(RREL), dim3(64), 0, stream>>>(rel_inds, preds, prior, Arel);

  auto gemm = [&](int epi, const _Float16* A1, const _Float16* A2, int ksplit,
                  const _Float16* W, const float* bias, void* C,
                  int M, int N, int K, int lda, int ldw, int ldc) {
    dim3 g((N + 127) / 128, (M + 127) / 128), b(256);
    if (epi == 0) gemm_wmma<0><<<g, b, 0, stream>>>(A1, A2, ksplit, W, bias, C, M, N, K, lda, ldw, ldc);
    if (epi == 1) gemm_wmma<1><<<g, b, 0, stream>>>(A1, A2, ksplit, W, bias, C, M, N, K, lda, ldw, ldc);
    if (epi == 2) gemm_wmma<2><<<g, b, 0, stream>>>(A1, A2, ksplit, W, bias, C, M, N, K, lda, ldw, ldc);
  };

  // ---- input projections ----
  gemm(0, ofm16, ofm16, 4096, opw16, obj_proj_b, of16, NOBJN, HH, 4096, 4096, 4096, HH);
  gemm(0, vr16,  vr16,  4096, rpw16, rel_proj_b, v16,  RREL,  HH, 4096, 4096, 4096, HH);

  // ---- build input_ggnn / initial hidden ----
  k_build_input<<<dim3(RREL), dim3(256), 0, stream>>>(of16, v16, rel_inds, inflat, flat);

  const size_t NBIG = BIGELEM;
  dim3 gBig((unsigned)((NBIG + 255) / 256)), b256(256);

  // ---- T GRU iterations ----
  for (int t = 0; t < 3; ++t) {
    k_xav<<<dim3(RREL), dim3(256), 0, stream>>>(flat, Arel, xav);
    gemm(0, flat, flat, HH, eq3u16, eq3u_b, gu,   BIGROWS, HH, HH, HH, HH, HH);
    gemm(0, xav,  xav,  HH, wzf,    eq3w_b, zpre, BIGROWS, HH, HH, HH, HH, HH);
    gemm(0, xav,  xav,  HH, wrf,    eq4w_b, rpre, BIGROWS, HH, HH, HH, HH, HH);
    gemm(0, xav,  xav,  HH, whf,    eq5w_b, hw,   BIGROWS, HH, HH, HH, HH, HH);
    k_zr<<<gBig, b256, 0, stream>>>(zpre, rpre, gu, flat, /*z*/zpre, /*rf*/xav, NBIG);
    gemm(0, xav,  xav,  HH, eq5u16, eq5u_b, gu,   BIGROWS, HH, HH, HH, HH, HH); // hu -> gu
    k_update<<<gBig, b256, 0, stream>>>(zpre, hw, gu, flat, NBIG);
  }

  // ---- out = relu([flat | input_flat] @ out_w^T + b) (split-K, K=1024) ----
  gemm(1, flat, inflat, HH, outw16, out_b, /*out*/zpre,
       BIGROWS, HH, 1024, HH, 1024, HH);

  // ---- rel_dists = out.reshape(R, NODE*OUT) @ cls_w^T + cls_b ----
  float* rel_dists = (float*)d_out + (size_t)NOBJN * NCLSN + NOBJN;
  gemm(2, zpre, zpre, NNODE * HH, clsw16, cls_b, rel_dists,
       RREL, NRELK, NNODE * HH, NNODE * HH, NNODE * HH, NRELK);

  // ---- pass-through outputs ----
  (void)hipMemcpyAsync(d_out, obj_logits, (size_t)NOBJN * NCLSN * sizeof(float),
                       hipMemcpyDeviceToDevice, stream);
  k_preds_out<<<dim3((NOBJN + 255) / 256), dim3(256), 0, stream>>>(
      preds, (float*)d_out + (size_t)NOBJN * NCLSN);
}
